// InterNet_26491358282084
// MI455X (gfx1250) — compile-verified, matrix operand order checked by reading, not run on hardware
//
#include <hip/hip_runtime.h>

typedef __bf16 bf16;
typedef __attribute__((ext_vector_type(16))) __bf16        v16bf;
typedef __attribute__((ext_vector_type(8)))  float         v8f;
typedef __attribute__((ext_vector_type(8)))  unsigned int  v8u;
typedef __attribute__((ext_vector_type(4)))  unsigned int  uint4v;

#define B_   128
#define N_   6
#define D_   256
#define P_   5
#define DS_  32
#define SP_  25            // P*P
#define NE_  (N_*D_*SP_)   // 38400 elements per batch sample (LN span)
#define S_   (B_*N_)       // 768 node samples
#define SR_  (B_*N_*(N_-1))// 3840 pair samples
#define QK_  (DS_*SP_)     // 800
#define FFK_ (D_*SP_)      // 6400
#define TEMP_ 28.284271247461902f
#define EPS_  1e-6f

// ---------- helpers ----------

__device__ __forceinline__ bf16 f2bf(float x) {
  union { float f; unsigned u; } a; a.f = x;
  unsigned r = (a.u + 0x7fffu + ((a.u >> 16) & 1u)) >> 16;
  union { unsigned short s; bf16 h; } b; b.s = (unsigned short)r;
  return b.h;
}

__device__ __forceinline__ v8f wmma_bf16(v16bf a, v16bf b, v8f c) {
  return __builtin_amdgcn_wmma_f32_16x16x32_bf16(false, a, false, b, (short)0, c, false, false);
}

// CDNA5 async global->LDS copy (tracked by ASYNCcnt), 16 bytes per lane.
__device__ __forceinline__ void async_copy_b128(unsigned lds_off, const void* gptr) {
  asm volatile("global_load_async_to_lds_b128 %0, %1, off"
               :: "v"(lds_off), "v"(gptr) : "memory");
}
__device__ __forceinline__ void wait_async0() {
  asm volatile("s_wait_asynccnt 0x0" ::: "memory");
}

// Fragment from LDS, row-major [r][k] with even row stride `ld` (elems).
// CDNA5 wave32 layout: lane l: r = l&15, h = l>>4; VGPR v holds K-pair at
// k0 = 2*(v&3) + 8h + 16*(v>>2)  ->  two ds_load_b128 per fragment.
__device__ __forceinline__ v16bf frag_from_lds(const bf16* base, int ld) {
  int lane = threadIdx.x & 31;
  int r = lane & 15, h = lane >> 4;
  const unsigned* p = (const unsigned*)(base + r * ld);
  v8u u;
#pragma unroll
  for (int v = 0; v < 8; ++v) {
    int k0 = 2 * (v & 3) + 8 * h + 16 * (v >> 2);
    u[v] = p[k0 >> 1];
  }
  return __builtin_bit_cast(v16bf, u);
}

// Fragment from pre-packed global layout: frag f = 32 lanes x 32 bytes,
// lane-contiguous -> two global_load_b128 per lane.
__device__ __forceinline__ v16bf frag_from_packed(const bf16* __restrict__ pack, long f) {
  int lane = threadIdx.x & 31;
  const uint4v* p = (const uint4v*)(pack + ((f << 5) + lane) * 16);
  uint4v a = p[0], b = p[1];
  v8u u;
#pragma unroll
  for (int i = 0; i < 4; ++i) { u[i] = a[i]; u[4 + i] = b[i]; }
  return __builtin_bit_cast(v16bf, u);
}

__device__ __forceinline__ void block_reduce2(float& a, float& b) {
  __shared__ float sa[256], sb[256];
  __syncthreads();
  sa[threadIdx.x] = a; sb[threadIdx.x] = b;
  __syncthreads();
  for (int o = 128; o > 0; o >>= 1) {
    if ((int)threadIdx.x < o) {
      sa[threadIdx.x] += sa[threadIdx.x + o];
      sb[threadIdx.x] += sb[threadIdx.x + o];
    }
    __syncthreads();
  }
  a = sa[0]; b = sb[0];
}

// ---------- elementwise cast ----------

__global__ __launch_bounds__(256) void cast_f32_bf16(const float* __restrict__ in,
                                                     bf16* __restrict__ out, long n) {
  long i = (long)blockIdx.x * 256 + threadIdx.x;
  if (i < n) out[i] = f2bf(in[i]);
}

// ---------- weight pack: fp32 [RO][K] -> per-fragment bf16 ----------
__global__ __launch_bounds__(256) void pack_frag(const float* __restrict__ src,
                                                 bf16* __restrict__ dst,
                                                 int RO, int K, int T) {
  int KT = K >> 5;
  long total = (long)T * KT * 32 * 16;
  long i = (long)blockIdx.x * 256 + threadIdx.x;
  if (i >= total) return;
  int e    = (int)(i & 15);
  int lane = (int)((i >> 4) & 31);
  long f   = i >> 9;
  int kt   = (int)(f % KT);
  int tile = (int)(f / KT);
  int r = tile * 16 + (lane & 15);
  int h = lane >> 4;
  int v = e >> 1, odd = e & 1;
  int k = kt * 32 + 2 * (v & 3) + 8 * h + 16 * (v >> 2) + odd;
  float val = (r < RO) ? src[(long)r * K + k] : 0.0f;
  dst[i] = f2bf(val);
}

// ---------- 1x1 attention conv (tiny, fp32 VALU) ----------
__global__ __launch_bounds__(256) void attn_conv1x1(const float* __restrict__ x,
                                                    const float* __restrict__ w,
                                                    const float* __restrict__ b,
                                                    float* __restrict__ s2d) {
  int s = blockIdx.x;
  const float* xin = x + (size_t)s * D_ * SP_;
  for (int o = threadIdx.x; o < DS_ * SP_; o += 256) {
    int ds = o / SP_, p = o % SP_;
    float acc = b[ds];
    const float* wr = w + (size_t)ds * D_;
    for (int ci = 0; ci < D_; ++ci) acc += wr[ci] * xin[ci * SP_ + p];
    s2d[(size_t)s * QK_ + o] = acc;
  }
}

// ---------- bf16 WMMA GEMM:  C[M][Nc] = A[M][K] * Bpack ----------
// Block tile 64M x 128N; 8 waves as 2(M) x 4(N); each wave 2x2 WMMA tiles
// (4 WMMAs / K-step from 2 A-frags + 2 B-frags -> 2x operand reuse).
// Double-buffered A panels staged with async global->LDS; 1 barrier / K-step.
__global__ __launch_bounds__(256) void gemm_bf16_wmma(const bf16* __restrict__ A,
                                                      const bf16* __restrict__ Bpack,
                                                      float* __restrict__ C,
                                                      int Nc, int K) {
  __shared__ alignas(16) bf16 As[2][64][40];
  int m0 = blockIdx.y * 64;
  int n0 = blockIdx.x * 128;
  int tid = threadIdx.x, wave = tid >> 5, lane = tid & 31;
  int mt2 = wave & 1;                // mtiles {2*mt2, 2*mt2+1}
  int nt4 = wave >> 1;               // ntiles {2*nt4, 2*nt4+1}
  int KT = K >> 5;
  long fb0 = (long)(blockIdx.x * 8 + nt4 * 2) * KT;
  long fb1 = fb0 + KT;

  v8f acc[2][2];
#pragma unroll
  for (int i = 0; i < 2; ++i)
#pragma unroll
    for (int j = 0; j < 2; ++j)
#pragma unroll
      for (int e = 0; e < 8; ++e) acc[i][j][e] = 0.0f;

  int am = tid >> 2;                 // 0..63
  int ak = (tid & 3) * 8;            // 0,8,16,24
  const bf16* gA = A + (size_t)(m0 + am) * K + ak;
  unsigned lds[2] = { (unsigned)(size_t)&As[0][am][ak],
                      (unsigned)(size_t)&As[1][am][ak] };

  async_copy_b128(lds[0], gA);       // stage kt=0
  wait_async0();
  __syncthreads();

  for (int kt = 0; kt < KT; ++kt) {
    int cur = kt & 1;
    if (kt + 1 < KT)                 // stage kt+1 while computing kt
      async_copy_b128(lds[cur ^ 1], gA + (size_t)(kt + 1) * 32);

    v16bf fa0 = frag_from_lds(&As[cur][(mt2 * 2 + 0) * 16][0], 40);
    v16bf fa1 = frag_from_lds(&As[cur][(mt2 * 2 + 1) * 16][0], 40);
    v16bf fbv0 = frag_from_packed(Bpack, fb0 + kt);
    v16bf fbv1 = frag_from_packed(Bpack, fb1 + kt);
    acc[0][0] = wmma_bf16(fa0, fbv0, acc[0][0]);
    acc[0][1] = wmma_bf16(fa0, fbv1, acc[0][1]);
    acc[1][0] = wmma_bf16(fa1, fbv0, acc[1][0]);
    acc[1][1] = wmma_bf16(fa1, fbv1, acc[1][1]);

    if (kt + 1 < KT) wait_async0();
    __syncthreads();
  }

  __builtin_amdgcn_s_wait_tensorcnt(0);   // CDNA5 wait path (TENSORcnt==0, harmless)

  int h = lane >> 4;
#pragma unroll
  for (int i = 0; i < 2; ++i) {
    int mrow0 = m0 + (mt2 * 2 + i) * 16 + h * 8;
#pragma unroll
    for (int j = 0; j < 2; ++j) {
      int ncol = n0 + (nt4 * 2 + j) * 16 + (lane & 15);
      if (ncol < Nc) {
#pragma unroll
        for (int v = 0; v < 8; ++v)
          C[(size_t)(mrow0 + v) * Nc + ncol] = acc[i][j][v];
      }
    }
  }
}

// ---------- implicit-GEMM 3x3 conv with WMMA ----------
// Async-staged input sample; double-buffered im2col panel (1 barrier / K-step).
// MODE 0: self -> atomicAdd(scale_self[s] * relu)      into Y[s]
// MODE 1: rel  -> atomicAdd(scale_rel[b,i,jj] * relu)  into Y[b*6+i] (concat-pair gather)
// MODE 2/3: aff/aggt -> store relu
template <int CIN, int MODE>
__global__ __launch_bounds__(256) void conv3x3_wmma(const bf16* __restrict__ Wpack,
                                                    const float* __restrict__ bias,
                                                    const bf16* __restrict__ Xbase,
                                                    const float* __restrict__ scale,
                                                    float* __restrict__ Y) {
  constexpr int K  = CIN * 9;
  constexpr int KT = K / 32;
  constexpr int XT = CIN * SP_;
  __shared__ alignas(16) bf16 Xs[XT];
  __shared__ alignas(16) bf16 Bt[2][32][40];   // n-major im2col: Bt[buf][col][k]

  int s = blockIdx.x;
  int tid = threadIdx.x;
  int wave = tid >> 5, lane = tid & 31;

  const bf16* src0;
  const bf16* src1;
  float sc = 1.0f;
  int outIdx;
  if (MODE == 1) {
    int b = s / 30, pi = s % 30;
    int i = pi / 5, jj = pi % 5;
    int j = (jj < i) ? jj : jj + 1;
    src0 = Xbase + (size_t)(b * N_ + i) * D_ * SP_;
    src1 = Xbase + (size_t)(b * N_ + j) * D_ * SP_;
    sc = scale[(b * N_ + i) * 5 + jj];
    outIdx = b * N_ + i;
  } else {
    src0 = Xbase + (size_t)s * CIN * SP_;
    src1 = src0;
    if (MODE == 0) sc = scale[s];
    outIdx = s;
  }

  // stage the whole input sample into LDS with async b128 copies
  for (int off = tid * 8; off < XT; off += 2048) {
    const bf16* sp = (MODE == 1 && off >= D_ * SP_) ? (src1 + (off - D_ * SP_))
                                                    : (src0 + off);
    async_copy_b128((unsigned)(size_t)(Xs + off), sp);
  }
  wait_async0();
  __syncthreads();                  // Xs visible to all waves

  auto build_panel = [&](int kt, bf16 (*BtBuf)[40]) {
    int k0 = kt * 32;
#pragma unroll
    for (int r = 0; r < 4; ++r) {   // 32x32 im2col panel, n-major
      int idx = r * 256 + tid;
      int col = idx >> 5, kk = idx & 31;
      bf16 v = f2bf(0.0f);
      if (col < SP_) {
        int kg = k0 + kk;
        int ci = kg / 9, t = kg - ci * 9;
        int dy = t / 3 - 1, dx = t - (t / 3) * 3 - 1;
        int py = col / 5 + dy, px = col % 5 + dx;
        if ((unsigned)py < 5u && (unsigned)px < 5u)
          v = Xs[ci * SP_ + py * 5 + px];
      }
      BtBuf[col][kk] = v;
    }
  };

  v8f acc[4];
#pragma unroll
  for (int t = 0; t < 4; ++t)
#pragma unroll
    for (int e = 0; e < 8; ++e) acc[t][e] = 0.0f;

  int ntile  = wave >> 2;           // 0..1  (cols [0,16) / [16,32))
  int mtbase = (wave & 3) * 4;      // 4 mtiles (64 Cout) per wave

  build_panel(0, Bt[0]);
  __syncthreads();

  for (int kt = 0; kt < KT; ++kt) {
    int cur = kt & 1;
    if (kt + 1 < KT) build_panel(kt + 1, Bt[cur ^ 1]);  // overlap with WMMAs

    v16bf fb = frag_from_lds(&Bt[cur][ntile * 16][0], 40);
#pragma unroll
    for (int t = 0; t < 4; ++t) {
      v16bf fa = frag_from_packed(Wpack, (long)(mtbase + t) * KT + kt);
      acc[t] = wmma_bf16(fa, fb, acc[t]);
    }
    __syncthreads();
  }

  int n = (lane & 15) + ntile * 16;
  int h = lane >> 4;
  if (n < SP_) {
#pragma unroll
    for (int t = 0; t < 4; ++t) {
#pragma unroll
      for (int v = 0; v < 8; ++v) {
        int m = (mtbase + t) * 16 + v + 8 * h;
        float val = fmaxf(acc[t][v] + bias[m], 0.0f);   // all four convs relu'd
        float* dst = Y + ((size_t)outIdx * D_ + m) * SP_ + n;
        if (MODE <= 1) atomicAdd(dst, sc * val);
        else           *dst = val;
      }
    }
  }
}

// ---------- attention softmax (per batch sample) ----------
__global__ __launch_bounds__(64) void attn_softmax(const float* __restrict__ q,
                                                   const float* __restrict__ k,
                                                   const float* __restrict__ valid,
                                                   float* __restrict__ scale_self,
                                                   float* __restrict__ scale_rel) {
  __shared__ float att[N_][N_];
  int b = blockIdx.x, tid = threadIdx.x;
  if (tid < N_ * N_) {
    int i = tid / N_, j = tid % N_;
    const float* qi = q + (size_t)(b * N_ + i) * QK_;
    const float* kj = k + (size_t)(b * N_ + j) * QK_;
    float acc = 0.f;
    for (int d = 0; d < QK_; ++d) acc += qi[d] * kj[d];
    acc = (acc / TEMP_) * valid[b * N_ + j];
    att[i][j] = (acc > 0.f) ? acc : -__builtin_inff();
  }
  __syncthreads();
  if (tid < N_) {
    int i = tid;
    float mx = -__builtin_inff();
    for (int j = 0; j < N_; ++j) mx = fmaxf(mx, att[i][j]);
    float e[N_], sum = 0.f;
    for (int j = 0; j < N_; ++j) {
      e[j] = (mx == -__builtin_inff()) ? 0.f : __expf(att[i][j] - mx);
      sum += e[j];
    }
    float inv = (sum > 0.f) ? ((float)N_ / sum) : 0.f;
    for (int j = 0; j < N_; ++j) e[j] *= inv;
    scale_self[b * N_ + i] = e[i];
    int jj = 0;
    for (int j = 0; j < N_; ++j)
      if (j != i) scale_rel[(b * N_ + i) * 5 + jj++] = 1.f + e[j];
  }
}

// ---------- double layernorm:  out = LN1( LN_aff(araw)*g0+b0 + s ) ----------
__global__ __launch_bounds__(256) void ln_double(const float* __restrict__ araw,
                                                 const float* __restrict__ s,
                                                 const float* __restrict__ g0,
                                                 const float* __restrict__ b0,
                                                 const float* __restrict__ g1,
                                                 const float* __restrict__ b1,
                                                 float* __restrict__ out) {
  int b = blockIdx.x;
  const float* ain = araw + (size_t)b * NE_;
  const float* sin_ = s + (size_t)b * NE_;
  float* o = out + (size_t)b * NE_;

  float sum = 0.f, sq = 0.f;
  for (int e = threadIdx.x; e < NE_; e += 256) { float v = ain[e]; sum += v; sq += v * v; }
  block_reduce2(sum, sq);
  float mu = sum / NE_;
  float rstd = rsqrtf(sq / NE_ - mu * mu + EPS_);

  float sum2 = 0.f, sq2 = 0.f;
  for (int e = threadIdx.x; e < NE_; e += 256) {
    float t = (ain[e] - mu) * rstd * g0[e] + b0[e] + sin_[e];
    o[e] = t; sum2 += t; sq2 += t * t;
  }
  block_reduce2(sum2, sq2);
  mu = sum2 / NE_;
  rstd = rsqrtf(sq2 / NE_ - mu * mu + EPS_);
  for (int e = threadIdx.x; e < NE_; e += 256)
    o[e] = (o[e] - mu) * rstd * g1[e] + b1[e];
}

// ---------- final:  out = LN2(ff + z) ----------
__global__ __launch_bounds__(256) void ln_final(const float* __restrict__ ff,
                                                const float* __restrict__ z,
                                                const float* __restrict__ g2,
                                                const float* __restrict__ b2,
                                                float* __restrict__ out) {
  int b = blockIdx.x;
  const float* fin = ff + (size_t)b * NE_;
  const float* zin = z + (size_t)b * NE_;
  float* o = out + (size_t)b * NE_;

  float sum = 0.f, sq = 0.f;
  for (int e = threadIdx.x; e < NE_; e += 256) {
    float t = fin[e] + zin[e];
    o[e] = t; sum += t; sq += t * t;
  }
  block_reduce2(sum, sq);
  float mu = sum / NE_;
  float rstd = rsqrtf(sq / NE_ - mu * mu + EPS_);
  for (int e = threadIdx.x; e < NE_; e += 256)
    o[e] = (o[e] - mu) * rstd * g2[e] + b2[e];
}

// ---------- host ----------

static inline int nblk(long n) { return (int)((n + 255) / 256); }

extern "C" void kernel_launch(void* const* d_in, const int* in_sizes, int n_in,
                              void* d_out, int out_size, void* d_ws, size_t ws_size,
                              hipStream_t stream) {
  const float* x      = (const float*)d_in[0];
  const float* valid  = (const float*)d_in[1];
  // d_in[2] g_idx: canonical deterministic pairs with weight 1 -> not needed
  const float* self_w = (const float*)d_in[3];
  const float* self_b = (const float*)d_in[4];
  const float* rel_w  = (const float*)d_in[5];
  const float* rel_b  = (const float*)d_in[6];
  const float* aff_w  = (const float*)d_in[7];
  const float* aff_b  = (const float*)d_in[8];
  const float* attn_w = (const float*)d_in[9];
  const float* attn_b = (const float*)d_in[10];
  const float* wq     = (const float*)d_in[11];
  const float* wk     = (const float*)d_in[12];
  const float* aggt_w = (const float*)d_in[13];
  const float* aggt_b = (const float*)d_in[14];
  const float* ff_w   = (const float*)d_in[15];
  const float* ln_aff_g = (const float*)d_in[16];
  const float* ln_aff_b = (const float*)d_in[17];
  const float* ln1_g  = (const float*)d_in[18];
  const float* ln1_b  = (const float*)d_in[19];
  const float* ln2_g  = (const float*)d_in[20];
  const float* ln2_b  = (const float*)d_in[21];
  float* out = (float*)d_out;

  char* wsp = (char*)d_ws;
  auto alloc = [&](size_t bytes) -> void* {
    void* p = (void*)wsp;
    wsp += (bytes + 255) & ~(size_t)255;
    return p;
  };

  const long NX = (long)B_ * N_ * D_ * SP_;           // 4,915,200

  const long NP_SELF = 16L * (D_ * 9 / 32) * 512;      //  589,824
  const long NP_REL  = 16L * (2 * D_ * 9 / 32) * 512;  // 1,179,648
  const long NP_QKW  = 56L * (QK_ / 32) * 512;         //   716,800 (N padded 800->896)
  const long NP_FF   = 400L * (FFK_ / 32) * 512;       // 40,960,000

  bf16*  x_bf     = (bf16*) alloc(NX * 2);
  bf16*  wp_self  = (bf16*) alloc(NP_SELF * 2);
  bf16*  wp_rel   = (bf16*) alloc(NP_REL * 2);
  bf16*  wp_aff   = (bf16*) alloc(NP_SELF * 2);
  bf16*  wp_aggt  = (bf16*) alloc(NP_SELF * 2);
  bf16*  wp_q     = (bf16*) alloc(NP_QKW * 2);
  bf16*  wp_k     = (bf16*) alloc(NP_QKW * 2);
  bf16*  wp_ff    = (bf16*) alloc(NP_FF * 2);
  float* s2d      = (float*)alloc((size_t)S_ * QK_ * 4);
  bf16*  s2d_bf   = (bf16*) alloc((size_t)S_ * QK_ * 2);
  float* qbuf     = (float*)alloc((size_t)S_ * QK_ * 4);
  float* kbuf     = (float*)alloc((size_t)S_ * QK_ * 4);
  float* sc_self  = (float*)alloc((size_t)S_ * 4);
  float* sc_rel   = (float*)alloc((size_t)S_ * 5 * 4);
  float* pred     = (float*)alloc(NX * 4);
  bf16*  pred_bf  = (bf16*) alloc(NX * 2);
  float* araw     = (float*)alloc(NX * 4);
  float* a2       = (float*)alloc(NX * 4);
  bf16*  a2_bf    = (bf16*) alloc(NX * 2);
  float* zbuf     = (float*)alloc(NX * 4);
  bf16*  z_bf     = (bf16*) alloc(NX * 2);
  float* ffbuf    = (float*)alloc(NX * 4);

  // 1) casts + fragment packing (weights read straight from their fp32 layouts)
  cast_f32_bf16<<<nblk(NX), 256, 0, stream>>>(x, x_bf, NX);
  pack_frag<<<nblk(NP_SELF), 256, 0, stream>>>(self_w, wp_self, D_, D_ * 9, 16);
  pack_frag<<<nblk(NP_REL),  256, 0, stream>>>(rel_w,  wp_rel,  D_, 2 * D_ * 9, 16);
  pack_frag<<<nblk(NP_SELF), 256, 0, stream>>>(aff_w,  wp_aff,  D_, D_ * 9, 16);
  pack_frag<<<nblk(NP_SELF), 256, 0, stream>>>(aggt_w, wp_aggt, D_, D_ * 9, 16);
  pack_frag<<<nblk(NP_QKW),  256, 0, stream>>>(wq,   wp_q,  QK_, QK_, 56);
  pack_frag<<<nblk(NP_QKW),  256, 0, stream>>>(wk,   wp_k,  QK_, QK_, 56);
  pack_frag<<<nblk(NP_FF),   256, 0, stream>>>(ff_w, wp_ff, FFK_, FFK_, 400);

  // 2) attention path: 1x1 conv -> q/k GEMM (WMMA) -> masked softmax
  attn_conv1x1<<<S_, 256, 0, stream>>>(x, attn_w, attn_b, s2d);
  cast_f32_bf16<<<nblk((long)S_ * QK_), 256, 0, stream>>>(s2d, s2d_bf, (long)S_ * QK_);
  {
    dim3 g((QK_ + 127) / 128, S_ / 64);
    gemm_bf16_wmma<<<g, 256, 0, stream>>>(s2d_bf, wp_q, qbuf, QK_, QK_);
    gemm_bf16_wmma<<<g, 256, 0, stream>>>(s2d_bf, wp_k, kbuf, QK_, QK_);
  }
  attn_softmax<<<B_, 64, 0, stream>>>(qbuf, kbuf, valid, sc_self, sc_rel);

  // 3) pred = self_wts*relu(self_conv(x)) + sum_pairs (1+rel_wts)*relu(rel_conv(pair))
  hipMemsetAsync(pred, 0, (size_t)NX * 4, stream);
  conv3x3_wmma<D_, 0><<<S_, 256, 0, stream>>>(wp_self, self_b, x_bf, sc_self, pred);
  conv3x3_wmma<2 * D_, 1><<<SR_, 256, 0, stream>>>(wp_rel, rel_b, x_bf, sc_rel, pred);

  // 4) a = LN1(LN_aff(relu(aff_conv(pred))) + x)
  cast_f32_bf16<<<nblk(NX), 256, 0, stream>>>(pred, pred_bf, NX);
  conv3x3_wmma<D_, 2><<<S_, 256, 0, stream>>>(wp_aff, aff_b, pred_bf, nullptr, araw);
  ln_double<<<B_, 256, 0, stream>>>(araw, x, ln_aff_g, ln_aff_b, ln1_g, ln1_b, a2);

  // 5) z = relu(aggt_conv(a));  ff = z @ ff_w^T;  out = LN2(ff + z)
  cast_f32_bf16<<<nblk(NX), 256, 0, stream>>>(a2, a2_bf, NX);
  conv3x3_wmma<D_, 3><<<S_, 256, 0, stream>>>(wp_aggt, aggt_b, a2_bf, nullptr, zbuf);
  cast_f32_bf16<<<nblk(NX), 256, 0, stream>>>(zbuf, z_bf, NX);
  {
    dim3 g(FFK_ / 128, S_ / 64);
    gemm_bf16_wmma<<<g, 256, 0, stream>>>(z_bf, wp_ff, ffbuf, FFK_, FFK_);
  }
  ln_final<<<B_, 256, 0, stream>>>(ffbuf, zbuf, ln2_g, ln2_b, out);

  (void)in_sizes; (void)n_in; (void)out_size; (void)ws_size;
}
